// MultiHeadAttention_25065429139985
// MI455X (gfx1250) — matrix operand order repaired and checked by reading, not compiled, not run on hardware
//
#include <hip/hip_runtime.h>
#include <hip/hip_bf16.h>

typedef __attribute__((ext_vector_type(16))) __bf16 v16bf;
typedef __attribute__((ext_vector_type(8)))  __bf16 v8bf;
typedef __attribute__((ext_vector_type(8)))  float  v8f;
typedef __attribute__((ext_vector_type(4)))  float  v4f;

#define B_  8
#define S_  1024
#define D_  1024
#define H_  16
#define HD_ 64

// ---------------------------------------------------------------------------
// helpers
// ---------------------------------------------------------------------------
__device__ __forceinline__ v8f wmma_bf16(v16bf a, v16bf b, v8f c) {
  return __builtin_amdgcn_wmma_f32_16x16x32_bf16(
      false, a, false, b, (short)0, c, false, false);
}

// IEEE maximumNumber directly on the VALU, no canonicalization pair
__device__ __forceinline__ float vmaxf(float a, float b) {
  float d;
  asm("v_max_num_f32 %0, %1, %2" : "=v"(d) : "v"(a), "v"(b));
  return d;
}

// A-fragment (16x32 bf16): lane<16 holds K {0..7,16..23}, lane>=16 {8..15,24..31}
__device__ __forceinline__ v16bf ldfragA(const __bf16* rowptr, int k0) {
  v8bf lo = *(const v8bf*)(rowptr + k0);
  v8bf hi = *(const v8bf*)(rowptr + k0 + 16);
  return __builtin_shufflevector(lo, hi, 0,1,2,3,4,5,6,7,8,9,10,11,12,13,14,15);
}

// B-fragment (32x16 bf16, K-contiguous per column): lane<16 K {0..7,16..23}, lane>=16 {8..15,24..31}
__device__ __forceinline__ v16bf ldfragB(const __bf16* colptr, int k0) {
  v8bf lo = *(const v8bf*)(colptr + k0);
  v8bf hi = *(const v8bf*)(colptr + k0 + 16);
  return __builtin_shufflevector(lo, hi, 0,1,2,3,4,5,6,7,8,9,10,11,12,13,14,15);
}

// 16-lane max reduction fully in the VALU via DPP
template <int CTRL>
__device__ __forceinline__ float dpp_max_step(float v) {
  union { float f; int i; } u, o;
  u.f = v;
  o.i = __builtin_amdgcn_update_dpp(u.i, u.i, CTRL, 0xf, 0xf, true);
  return vmaxf(v, o.f);
}
__device__ __forceinline__ float hmax16_dpp(float v) {
  v = dpp_max_step<0xB1>(v);    // quad_perm(1,0,3,2)
  v = dpp_max_step<0x4E>(v);    // quad_perm(2,3,0,1)
  v = dpp_max_step<0x141>(v);   // row_half_mirror
  v = dpp_max_step<0x140>(v);   // row_mirror
  return v;
}

__device__ __forceinline__ void wave_fence() {
  __builtin_amdgcn_fence(__ATOMIC_ACQ_REL, "wavefront");
  __builtin_amdgcn_wave_barrier();
}

// ---------------------------------------------------------------------------
// Weight transpose + convert: in f32 [R][C] -> out bf16 [C][R], 64x64 LDS tiles
// ---------------------------------------------------------------------------
__global__ __launch_bounds__(256)
void mha_wcvt_t(const float* __restrict__ in, __bf16* __restrict__ out,
                int R, int C) {
  __shared__ __bf16 tl[64][72];
  const int r0 = blockIdx.y * 64, c0 = blockIdx.x * 64;
  const int t = threadIdx.x;
  {
    int rr = t >> 2, cc = (t & 3) * 16;
    const float* src = in + (size_t)(r0 + rr) * C + c0 + cc;
    v8bf lo, hi;
#pragma unroll
    for (int e = 0; e < 8; ++e) { lo[e] = (__bf16)src[e]; hi[e] = (__bf16)src[e + 8]; }
    *(v8bf*)&tl[rr][cc] = lo;
    *(v8bf*)&tl[rr][cc + 8] = hi;
  }
  __syncthreads();
  {
    int cr = t >> 2, rc = (t & 3) * 16;
    v8bf o0, o1;
#pragma unroll
    for (int e = 0; e < 8; ++e) { o0[e] = tl[rc + e][cr]; o1[e] = tl[rc + 8 + e][cr]; }
    __bf16* dst = out + (size_t)(c0 + cr) * R + r0 + rc;
    *(v8bf*)dst = o0;
    *(v8bf*)(dst + 8) = o1;
  }
}

// ---------------------------------------------------------------------------
// V transpose (bf16): vt[b][h][d][s] = kv[b][s][D + h*HD + d], 64x64 LDS tiles
// ---------------------------------------------------------------------------
__global__ __launch_bounds__(256)
void mha_vtrans(const __bf16* __restrict__ kv, __bf16* __restrict__ vt) {
  __shared__ __bf16 tl[64][72];
  const int st = blockIdx.x * 64;
  const int bh = blockIdx.y;
  const int b = bh >> 4, h = bh & 15;
  const int t = threadIdx.x;
  {
    int sr = t >> 2, dc = (t & 3) * 16;
    const __bf16* src = kv + ((size_t)b * S_ + st + sr) * (2 * D_) + D_ + h * HD_ + dc;
    *(v8bf*)&tl[sr][dc] = *(const v8bf*)src;
    *(v8bf*)&tl[sr][dc + 8] = *(const v8bf*)(src + 8);
  }
  __syncthreads();
  {
    int dr = t >> 2, sc = (t & 3) * 16;
    v8bf o0, o1;
#pragma unroll
    for (int e = 0; e < 8; ++e) { o0[e] = tl[sc + e][dr]; o1[e] = tl[sc + 8 + e][dr]; }
    __bf16* dst = vt + ((size_t)bh * HD_ + dr) * S_ + st + sc;
    *(v8bf*)dst = o0;
    *(v8bf*)(dst + 8) = o1;
  }
}

// ---------------------------------------------------------------------------
// Tiled GEMM, software pipelined with double-buffered LDS:
//   C[M,N] = A[M,K] * Bt[N,K]^T + bias ; A is f32 or bf16.
// Block tile 128x128, 256 threads (8 waves), 8 WMMA tiles/wave, K-step 32.
// Next tile's global loads are in flight during current tile's WMMAs.
// ---------------------------------------------------------------------------
template <typename AT, int STOREF32>
__global__ __launch_bounds__(256)
void mha_gemm(const AT* __restrict__ A, const __bf16* __restrict__ Bt,
              const float* __restrict__ bias, void* __restrict__ Cout,
              int M, int N, int K, float scale) {
  __shared__ __bf16 Alds[2][128 * 40];
  __shared__ __bf16 Blds[2][128 * 40];

  const int bm = blockIdx.y * 128;
  const int bn = blockIdx.x * 128;
  const int t = threadIdx.x;
  const int w = t >> 5;
  const int lane = t & 31;
  const int half = lane >> 4;
  const int l16 = lane & 15;
  const int wm = (w & 3) * 32;
  const int wn = (w >> 2) * 64;

  const int sr = t >> 1;
  const int skc = (t & 1) * 16;

  v4f  af32[4];
  v8bf ah16[2];
  v8bf breg[2];

  auto loadNext = [&](int kk) {
    const AT* srcA = A + (size_t)(bm + sr) * K + kk + skc;
    if constexpr (sizeof(AT) == 4) {
      af32[0] = *(const v4f*)(srcA);
      af32[1] = *(const v4f*)(srcA + 4);
      af32[2] = *(const v4f*)(srcA + 8);
      af32[3] = *(const v4f*)(srcA + 12);
    } else {
      ah16[0] = *(const v8bf*)(srcA);
      ah16[1] = *(const v8bf*)(srcA + 8);
    }
    const __bf16* srcB = Bt + (size_t)(bn + sr) * K + kk + skc;
    breg[0] = *(const v8bf*)(srcB);
    breg[1] = *(const v8bf*)(srcB + 8);
  };

  auto storeBuf = [&](int bi) {
    v8bf lo, hi;
    if constexpr (sizeof(AT) == 4) {
#pragma unroll
      for (int e = 0; e < 8; ++e) {
        lo[e] = (__bf16)af32[e >> 2][e & 3];
        hi[e] = (__bf16)af32[2 + (e >> 2)][e & 3];
      }
    } else {
      lo = ah16[0];
      hi = ah16[1];
    }
    *(v8bf*)(Alds[bi] + sr * 40 + skc) = lo;
    *(v8bf*)(Alds[bi] + sr * 40 + skc + 8) = hi;
    *(v8bf*)(Blds[bi] + sr * 40 + skc) = breg[0];
    *(v8bf*)(Blds[bi] + sr * 40 + skc + 8) = breg[1];
  };

  v8f acc[2][4];
#pragma unroll
  for (int i = 0; i < 2; ++i)
#pragma unroll
    for (int j = 0; j < 4; ++j) acc[i][j] = (v8f){};

  const int NK = K / 32;
  loadNext(0);
  storeBuf(0);
  __syncthreads();

  for (int kb = 0; kb < NK; ++kb) {
    const int cur = kb & 1;
    const bool hasNext = (kb + 1 < NK);
    if (hasNext) loadNext((kb + 1) * 32);   // in flight during the WMMAs below

    v16bf af[2], bfv[4];
#pragma unroll
    for (int i = 0; i < 2; ++i)
      af[i] = ldfragA(Alds[cur] + (wm + i * 16 + l16) * 40, half * 8);
#pragma unroll
    for (int j = 0; j < 4; ++j)
      bfv[j] = ldfragB(Blds[cur] + (wn + j * 16 + l16) * 40, half * 8);
#pragma unroll
    for (int i = 0; i < 2; ++i)
#pragma unroll
      for (int j = 0; j < 4; ++j)
        acc[i][j] = wmma_bf16(af[i], bfv[j], acc[i][j]);

    if (hasNext) storeBuf(1 - cur);
    __syncthreads();
  }

#pragma unroll
  for (int j = 0; j < 4; ++j) {
    const int col = bn + wn + j * 16 + l16;
    const float bv = bias[col];
#pragma unroll
    for (int i = 0; i < 2; ++i) {
#pragma unroll
      for (int r = 0; r < 8; ++r) {
        const int row = bm + wm + i * 16 + r + half * 8;
        const float o = (acc[i][j][r] + bv) * scale;
        if (STOREF32)
          ((float*)Cout)[(size_t)row * N + col] = o;
        else
          ((__bf16*)Cout)[(size_t)row * N + col] = (__bf16)o;
      }
    }
  }
}

// ---------------------------------------------------------------------------
// Flash attention: 4 waves/block, each wave owns (b, h, 32 query rows).
// Scores arrive pre-scaled by log2(e)/sqrt(HD) -> softmax runs in exp2 domain.
// Denominator = 5th WMMA accumulator column (P x ones); row-max via DPP.
// 18 WMMAs per 32-key chunk per wave.
// ---------------------------------------------------------------------------
__global__ __launch_bounds__(128)
void mha_attn(const __bf16* __restrict__ qbuf, const __bf16* __restrict__ kvbuf,
              const __bf16* __restrict__ vtbuf, __bf16* __restrict__ heads) {
  __shared__ __bf16 ptile[4][32 * 40];

  const int w = threadIdx.x >> 5;
  const int lane = threadIdx.x & 31;
  const int idx = blockIdx.x * 4 + w;     // B*H*(S/32) = 4096
  const int qt = idx & 31;
  const int h  = (idx >> 5) & (H_ - 1);
  const int b  = idx >> 9;
  const int half = lane >> 4;
  const int l16 = lane & 15;
  __bf16* pt = &ptile[w][0];

  const int ak0 = half * 8;
  const int kb0 = half * 8;

  v16bf qa[2][2];
#pragma unroll
  for (int mt = 0; mt < 2; ++mt) {
    const __bf16* qrow =
        qbuf + ((size_t)b * S_ + qt * 32 + mt * 16 + l16) * D_ + h * HD_;
    qa[mt][0] = ldfragA(qrow, ak0);
    qa[mt][1] = ldfragA(qrow, ak0 + 32);
  }

  v16bf vone;
#pragma unroll
  for (int e = 0; e < 16; ++e) vone[e] = (__bf16)1.0f;

  float m_r[2][8];
  v8f acc[2][5];                          // [mt][0..3]=O columns, [4]=l (P*ones)
#pragma unroll
  for (int mt = 0; mt < 2; ++mt) {
#pragma unroll
    for (int r = 0; r < 8; ++r) m_r[mt][r] = -1.0e30f;
#pragma unroll
    for (int c = 0; c < 5; ++c) acc[mt][c] = (v8f){};
  }

  for (int kt = 0; kt < S_ / 32; ++kt) {
    // ---- scores for the full 32-key chunk (combined max is exact) ----
    v8f s[2][2];
#pragma unroll
    for (int jj = 0; jj < 2; ++jj) {
      const int krow = kt * 32 + jj * 16 + l16;
      const __bf16* kptr = kvbuf + ((size_t)b * S_ + krow) * (2 * D_) + h * HD_;
      const v16bf kbf0 = ldfragB(kptr, kb0);
      const v16bf kbf1 = ldfragB(kptr + 32, kb0);
#pragma unroll
      for (int mt = 0; mt < 2; ++mt) {
        v8f sv = {};
        sv = wmma_bf16(qa[mt][0], kbf0, sv);
        sv = wmma_bf16(qa[mt][1], kbf1, sv);
        s[mt][jj] = sv;
      }
    }

    // ---- online softmax in exp2 domain (asm v_max: no canonicalize pairs) ----
#pragma unroll
    for (int mt = 0; mt < 2; ++mt) {
#pragma unroll
      for (int r = 0; r < 8; ++r) {
        float smax = hmax16_dpp(vmaxf(s[mt][0][r], s[mt][1][r]));
        const float mn = vmaxf(m_r[mt][r], smax);
        const float corr = exp2f(m_r[mt][r] - mn);
        m_r[mt][r] = mn;
        const float p0 = exp2f(s[mt][0][r] - mn);
        const float p1 = exp2f(s[mt][1][r] - mn);
#pragma unroll
        for (int c = 0; c < 5; ++c) acc[mt][c][r] *= corr;
        const int prow = (mt * 16 + r + half * 8) * 40;
        pt[prow + l16] = (__bf16)p0;
        pt[prow + 16 + l16] = (__bf16)p1;
      }
    }
    wave_fence();                         // P stores -> P fragment loads

    // ---- P x V (and P x ones for the denominator) ----
    const v16bf pa0 = ldfragA(pt + l16 * 40, ak0);
    const v16bf pa1 = ldfragA(pt + (16 + l16) * 40, ak0);
    const int kbase = kt * 32;
#pragma unroll
    for (int c = 0; c < 4; ++c) {
      const __bf16* vptr =
          vtbuf + (((size_t)(b * H_ + h)) * HD_ + c * 16 + l16) * S_ + kbase;
      const v16bf vb = ldfragB(vptr, kb0);
      acc[0][c] = wmma_bf16(pa0, vb, acc[0][c]);
      acc[1][c] = wmma_bf16(pa1, vb, acc[1][c]);
    }
    acc[0][4] = wmma_bf16(pa0, vone, acc[0][4]);
    acc[1][4] = wmma_bf16(pa1, vone, acc[1][4]);

    if (kt + 1 < S_ / 32) {
      __builtin_prefetch(
          kvbuf + ((size_t)b * S_ + (kt + 1) * 32 + l16) * (2 * D_) + h * HD_, 0, 1);
    }
    wave_fence();                         // P loads done before next overwrite
  }

#pragma unroll
  for (int mt = 0; mt < 2; ++mt) {
#pragma unroll
    for (int r = 0; r < 8; ++r) {
      const float inv_l = 1.0f / acc[mt][4][r];
      const int row = qt * 32 + mt * 16 + r + half * 8;
#pragma unroll
      for (int c = 0; c < 4; ++c) {
        const int col = h * HD_ + c * 16 + l16;
        heads[((size_t)b * S_ + row) * D_ + col] = (__bf16)(acc[mt][c][r] * inv_l);
      }
    }
  }
}

// ---------------------------------------------------------------------------
// launch
// ---------------------------------------------------------------------------
extern "C" void kernel_launch(void* const* d_in, const int* in_sizes, int n_in,
                              void* d_out, int out_size, void* d_ws,
                              size_t ws_size, hipStream_t stream) {
  const float* query = (const float*)d_in[0];
  const float* value = (const float*)d_in[1];
  const float* Wq    = (const float*)d_in[2];
  const float* bq    = (const float*)d_in[3];
  const float* Wkv   = (const float*)d_in[4];
  const float* bkv   = (const float*)d_in[5];
  const float* Wo    = (const float*)d_in[6];
  const float* bo    = (const float*)d_in[7];
  float* out = (float*)d_out;

  const int nAct = B_ * S_ * D_;
  const int nKv  = B_ * S_ * 2 * D_;

  size_t off = 0;
  auto carve = [&](size_t bytes) -> char* {
    char* p = (char*)d_ws + off;
    off += (bytes + 255) & ~(size_t)255;
    return p;
  };
  __bf16* wqT   = (__bf16*)carve((size_t)D_ * D_ * 2);
  __bf16* wkvT  = (__bf16*)carve((size_t)2 * D_ * D_ * 2);
  __bf16* woT   = (__bf16*)carve((size_t)D_ * D_ * 2);
  __bf16* qbuf  = (__bf16*)carve((size_t)nAct * 2);
  __bf16* kvbuf = (__bf16*)carve((size_t)nKv * 2);
  __bf16* vtbuf = (__bf16*)carve((size_t)nAct * 2);
  __bf16* hbuf  = (__bf16*)carve((size_t)nAct * 2);

  mha_wcvt_t<<<dim3(D_ / 64, D_ / 64), 256, 0, stream>>>(Wq, wqT, D_, D_);
  mha_wcvt_t<<<dim3(2 * D_ / 64, D_ / 64), 256, 0, stream>>>(Wkv, wkvT, D_, 2 * D_);
  mha_wcvt_t<<<dim3(D_ / 64, D_ / 64), 256, 0, stream>>>(Wo, woT, D_, D_);

  const int M = B_ * S_;
  // q-proj folds softmax scale AND log2(e) for the exp2-domain softmax
  const float qscale = 0.125f * 1.4426950408889634f;
  mha_gemm<float, 0><<<dim3(D_ / 128, M / 128), 256, 0, stream>>>(
      query, wqT, bq, qbuf, M, D_, D_, qscale);
  mha_gemm<float, 0><<<dim3(2 * D_ / 128, M / 128), 256, 0, stream>>>(
      value, wkvT, bkv, kvbuf, M, 2 * D_, D_, 1.0f);
  mha_vtrans<<<dim3(S_ / 64, B_ * H_), 256, 0, stream>>>(kvbuf, vtbuf);
  mha_attn<<<(B_ * H_ * (S_ / 32)) / 4, 128, 0, stream>>>(qbuf, kvbuf, vtbuf, hbuf);
  mha_gemm<__bf16, 1><<<dim3(D_ / 128, M / 128), 256, 0, stream>>>(
      hbuf, woT, bo, out, M, D_, D_, 1.0f);
}
